// SimplifiedTHGFD_76312978915563
// MI455X (gfx1250) — matrix operand order repaired and checked by reading, not compiled, not run on hardware
//
#include <hip/hip_runtime.h>
#include <hip/hip_bf16.h>

#define N_TX    100000
#define N_USER  20000
#define N_MERCH 5000
#define NEDGE   400000
#define F_IN    32
#define HDIM    256

typedef __attribute__((ext_vector_type(16))) __bf16 v16bf;
typedef __attribute__((ext_vector_type(8)))  float  v8f;

union Frag16 { uint4 q[2]; v16bf v; };

#define WT_PAD 272                       // 256+16 elems -> 544B row stride, 16B-aligned runs
#define LDS_TILE (64 * WT_PAD)           // elements per weight tile
#define DYN_LDS  (2 * LDS_TILE * 2 + 64 * 4)   // two bf16 tiles + f32 bias row

// ---------------------------------------------------------------------------
// Input projection: xt_bf = bf16(x_tx @ Wp + bp)
// ---------------------------------------------------------------------------
__global__ __launch_bounds__(256) void proj_kernel(
    const float* __restrict__ x, const float* __restrict__ Wp,
    const float* __restrict__ bp, __bf16* __restrict__ out)
{
    __shared__ float xs[F_IN];
    const int row = blockIdx.x;
    const int tid = threadIdx.x;
    if (tid < F_IN) xs[tid] = x[(size_t)row * F_IN + tid];
    __syncthreads();
    float s = bp[tid];
#pragma unroll
    for (int k = 0; k < F_IN; ++k)
        s = fmaf(xs[k], Wp[k * HDIM + tid], s);
    out[(size_t)row * HDIM + tid] = (__bf16)s;
}

// ---------------------------------------------------------------------------
// Elementwise utilities
// ---------------------------------------------------------------------------
__global__ void zero_kernel(float* __restrict__ p, long n)
{
    long i = (long)blockIdx.x * blockDim.x + threadIdx.x;
    const long stride = (long)gridDim.x * blockDim.x;
    for (; i < n; i += stride) p[i] = 0.0f;
}

__global__ void f32_to_bf16_kernel(const float* __restrict__ in, __bf16* __restrict__ out, long n)
{
    long i = (long)blockIdx.x * blockDim.x + threadIdx.x;
    const long stride = (long)gridDim.x * blockDim.x;
    for (; i < n; i += stride) out[i] = (__bf16)in[i];
}

// mean_bf[i] = bf16( mean[i] / max(cnt[row],1) )
__global__ void div_cnt_kernel(const float* __restrict__ mean, const float* __restrict__ cnt,
                               __bf16* __restrict__ out, long n)
{
    long i = (long)blockIdx.x * blockDim.x + threadIdx.x;
    const long stride = (long)gridDim.x * blockDim.x;
    for (; i < n; i += stride)
        out[i] = (__bf16)(mean[i] / fmaxf(cnt[i >> 8], 1.0f));   // HDIM == 256
}

// x_bf[i] = bf16( relu(scale * acc[i]) )
__global__ void relu_scale_kernel(const float* __restrict__ in, __bf16* __restrict__ out,
                                  float scale, long n)
{
    long i = (long)blockIdx.x * blockDim.x + threadIdx.x;
    const long stride = (long)gridDim.x * blockDim.x;
    for (; i < n; i += stride)
        out[i] = (__bf16)fmaxf(in[i] * scale, 0.0f);
}

// ---------------------------------------------------------------------------
// Edge scatter: one wave per edge; gather bf16 source row, f32 atomics.
// ---------------------------------------------------------------------------
__global__ __launch_bounds__(256) void scatter_kernel(
    const __bf16* __restrict__ xsrc, const int* __restrict__ src,
    const int* __restrict__ dst, float* __restrict__ acc,
    float* __restrict__ cnt, int nedge)
{
    const int e = blockIdx.x * 8 + (threadIdx.x >> 5);
    if (e >= nedge) return;
    const int lane = threadIdx.x & 31;
    const int s = src[e];
    const int d = dst[e];
    const __bf16* srow = xsrc + (size_t)s * HDIM;
    float* drow = acc + (size_t)d * HDIM;
#pragma unroll
    for (int i = 0; i < HDIM / 32; ++i)
        atomicAdd(&drow[lane + i * 32], (float)srow[lane + i * 32]);
    if (lane == 0) atomicAdd(&cnt[d], 1.0f);
}

// ---------------------------------------------------------------------------
// Fused 2-term WMMA GEMM (compile-time specialized):
//   C[M,Nt] (+)= A0@W0 ( + A1@W1 ) + bias (relu)
// A streams bf16 (raw 16B fragment loads); weight slices transposed in dyn LDS.
// Block: 256 thr = 8 waves; tile 128(M) x 64(N); full K=256, fully unrolled.
// ---------------------------------------------------------------------------
template <bool TWO, bool ACCUM, bool RELU>
__global__ __launch_bounds__(256) void gemm2_wmma_kernel(
    const __bf16* __restrict__ A0, const float* __restrict__ W0,
    const __bf16* __restrict__ A1, const float* __restrict__ W1,
    const float* __restrict__ bias, float* __restrict__ C,
    int M, int Nt)
{
    extern __shared__ char smem[];
    __bf16* Wt0 = (__bf16*)smem;                    // [64][WT_PAD]
    __bf16* Wt1 = Wt0 + LDS_TILE;                   // [64][WT_PAD]
    float*  bsm = (float*)(Wt1 + LDS_TILE);         // [64]

    const int tid  = threadIdx.x;
    const int n0   = blockIdx.y * 64;
    const int row0 = blockIdx.x * 128;

    // Stage W^T slices: W[k][n0+n] -> Wt[n*WT_PAD + k], f32 -> bf16 (coalesced reads)
    for (int idx = tid; idx < 64 * 256; idx += 256) {
        const int n = idx & 63;
        const int k = idx >> 6;
        Wt0[n * WT_PAD + k] = (__bf16)W0[(size_t)k * Nt + (n0 + n)];
        if (TWO)
            Wt1[n * WT_PAD + k] = (__bf16)W1[(size_t)k * Nt + (n0 + n)];
    }
    if (tid < 64) bsm[tid] = bias ? bias[n0 + tid] : 0.0f;
    __syncthreads();

    const int wave = tid >> 5;
    const int lane = tid & 31;
    const int half = lane >> 4;
    const int r    = lane & 15;

    const int arow_i = row0 + wave * 16 + r;
    const int arow_c = arow_i < M ? arow_i : (M - 1);     // clamp; masked at store
    const __bf16* __restrict__ ar0 = A0 + (size_t)arow_c * HDIM;
    const __bf16* __restrict__ ar1 = A1 + (size_t)arow_c * HDIM;

    v8f acc0 = {}, acc1 = {}, acc2 = {}, acc3 = {};

#pragma unroll
    for (int k0 = 0; k0 < 256; k0 += 32) {
        const int kb0 = k0 + half * 16;   // A layout: lanes 0-15 K 0-15, lanes 16-31 K 16-31
        const int kb  = k0 + half * 16;   // B layout: lanes 0-15 K 0-15, lanes 16-31 K 16-31

        // ---- term 0
        {
            Frag16 a;
            a.q[0] = *(const uint4*)(ar0 + kb0);
            a.q[1] = *(const uint4*)(ar0 + kb0 + 8);
            Frag16 b0, b1, b2, b3;
            const __bf16* wp0 = &Wt0[(r)      * WT_PAD + kb];
            const __bf16* wp1 = &Wt0[(16 + r) * WT_PAD + kb];
            const __bf16* wp2 = &Wt0[(32 + r) * WT_PAD + kb];
            const __bf16* wp3 = &Wt0[(48 + r) * WT_PAD + kb];
            b0.q[0] = *(const uint4*)(wp0); b0.q[1] = *(const uint4*)(wp0 + 8);
            b1.q[0] = *(const uint4*)(wp1); b1.q[1] = *(const uint4*)(wp1 + 8);
            b2.q[0] = *(const uint4*)(wp2); b2.q[1] = *(const uint4*)(wp2 + 8);
            b3.q[0] = *(const uint4*)(wp3); b3.q[1] = *(const uint4*)(wp3 + 8);
            acc0 = __builtin_amdgcn_wmma_f32_16x16x32_bf16(false, a.v, false, b0.v, (short)0, acc0, false, false);
            acc1 = __builtin_amdgcn_wmma_f32_16x16x32_bf16(false, a.v, false, b1.v, (short)0, acc1, false, false);
            acc2 = __builtin_amdgcn_wmma_f32_16x16x32_bf16(false, a.v, false, b2.v, (short)0, acc2, false, false);
            acc3 = __builtin_amdgcn_wmma_f32_16x16x32_bf16(false, a.v, false, b3.v, (short)0, acc3, false, false);
        }
        // ---- term 1 (fused second SAGE term)
        if (TWO) {
            Frag16 a;
            a.q[0] = *(const uint4*)(ar1 + kb0);
            a.q[1] = *(const uint4*)(ar1 + kb0 + 8);
            Frag16 b0, b1, b2, b3;
            const __bf16* wp0 = &Wt1[(r)      * WT_PAD + kb];
            const __bf16* wp1 = &Wt1[(16 + r) * WT_PAD + kb];
            const __bf16* wp2 = &Wt1[(32 + r) * WT_PAD + kb];
            const __bf16* wp3 = &Wt1[(48 + r) * WT_PAD + kb];
            b0.q[0] = *(const uint4*)(wp0); b0.q[1] = *(const uint4*)(wp0 + 8);
            b1.q[0] = *(const uint4*)(wp1); b1.q[1] = *(const uint4*)(wp1 + 8);
            b2.q[0] = *(const uint4*)(wp2); b2.q[1] = *(const uint4*)(wp2 + 8);
            b3.q[0] = *(const uint4*)(wp3); b3.q[1] = *(const uint4*)(wp3 + 8);
            acc0 = __builtin_amdgcn_wmma_f32_16x16x32_bf16(false, a.v, false, b0.v, (short)0, acc0, false, false);
            acc1 = __builtin_amdgcn_wmma_f32_16x16x32_bf16(false, a.v, false, b1.v, (short)0, acc1, false, false);
            acc2 = __builtin_amdgcn_wmma_f32_16x16x32_bf16(false, a.v, false, b2.v, (short)0, acc2, false, false);
            acc3 = __builtin_amdgcn_wmma_f32_16x16x32_bf16(false, a.v, false, b3.v, (short)0, acc3, false, false);
        }
    }

    // ---- Epilogue. C/D layout: VGPR v -> M = v + half*8, N = r (per 16-col subtile).
    const int obase = row0 + wave * 16 + half * 8;
    const float bv0 = bsm[r], bv1 = bsm[16 + r], bv2 = bsm[32 + r], bv3 = bsm[48 + r];
#pragma unroll
    for (int v = 0; v < 8; ++v) {
        const int orow = obase + v;
        if (orow < M) {
            float* cp = C + (size_t)orow * Nt + n0 + r;
            float v0 = acc0[v] + bv0;
            float v1 = acc1[v] + bv1;
            float v2 = acc2[v] + bv2;
            float v3 = acc3[v] + bv3;
            if (ACCUM) { v0 += cp[0]; v1 += cp[16]; v2 += cp[32]; v3 += cp[48]; }
            if (RELU) {
                v0 = fmaxf(v0, 0.0f); v1 = fmaxf(v1, 0.0f);
                v2 = fmaxf(v2, 0.0f); v3 = fmaxf(v3, 0.0f);
            }
            cp[0] = v0; cp[16] = v1; cp[32] = v2; cp[48] = v3;
        }
    }
}

// ---------------------------------------------------------------------------
// Head output: out[row] = dot(z[row,:128], w) + b   (one wave per row)
// ---------------------------------------------------------------------------
__global__ __launch_bounds__(256) void head_out_kernel(
    const float* __restrict__ z, const float* __restrict__ w,
    const float* __restrict__ b, float* __restrict__ out, int M)
{
    const int row = blockIdx.x * 8 + (threadIdx.x >> 5);
    if (row >= M) return;
    const int lane = threadIdx.x & 31;
    const float* zr = z + (size_t)row * 128;
    float s = 0.0f;
#pragma unroll
    for (int i = 0; i < 4; ++i)
        s = fmaf(zr[lane + i * 32], w[lane + i * 32], s);
#pragma unroll
    for (int off = 16; off > 0; off >>= 1)
        s += __shfl_down(s, off, 32);
    if (lane == 0) out[row] = s + b[0];
}

// ---------------------------------------------------------------------------
// Host-side helpers
// ---------------------------------------------------------------------------
static inline unsigned gblocks(long n) {
    long b = (n + 255) / 256;
    return (unsigned)(b < 1 ? 1 : b);
}

// Fused 2-term, init:  C = A0@W0 + A1@W1 + bias
static void launch_gemm_fused_init(hipStream_t stream,
                                   const __bf16* A0, const float* W0,
                                   const __bf16* A1, const float* W1,
                                   const float* bias, float* C, int M, int Nt)
{
    dim3 grid((M + 127) / 128, Nt / 64);
    gemm2_wmma_kernel<true, false, false><<<grid, 256, DYN_LDS, stream>>>(A0, W0, A1, W1, bias, C, M, Nt);
}

// Fused 2-term, accumulate:  C += A0@W0 + A1@W1 + bias
static void launch_gemm_fused_acc(hipStream_t stream,
                                  const __bf16* A0, const float* W0,
                                  const __bf16* A1, const float* W1,
                                  const float* bias, float* C, int M, int Nt)
{
    dim3 grid((M + 127) / 128, Nt / 64);
    gemm2_wmma_kernel<true, true, false><<<grid, 256, DYN_LDS, stream>>>(A0, W0, A1, W1, bias, C, M, Nt);
}

// Single-term + relu (heads):  C = relu(A0@W0 + bias)
static void launch_gemm_relu(hipStream_t stream,
                             const __bf16* A0, const float* W0,
                             const float* bias, float* C, int M, int Nt)
{
    dim3 grid((M + 127) / 128, Nt / 64);
    gemm2_wmma_kernel<false, false, true><<<grid, 256, DYN_LDS, stream>>>(A0, W0, A0, W0, bias, C, M, Nt);
}

// seg-mean of xsrc over (src,dst) edges into bf16 out_bf  (uses f32 scratch mean+cnt)
static void launch_segmean(hipStream_t stream, const __bf16* xsrc,
                           const int* src, const int* dst, int ndst,
                           float* mean, float* cnt, __bf16* out_bf)
{
    const long nm = (long)ndst * HDIM;
    zero_kernel<<<gblocks(nm), 256, 0, stream>>>(mean, nm);
    zero_kernel<<<gblocks(ndst), 256, 0, stream>>>(cnt, ndst);
    scatter_kernel<<<(NEDGE + 7) / 8, 256, 0, stream>>>(xsrc, src, dst, mean, cnt, NEDGE);
    div_cnt_kernel<<<gblocks(nm), 256, 0, stream>>>(mean, cnt, out_bf, nm);
}

extern "C" void kernel_launch(void* const* d_in, const int* in_sizes, int n_in,
                              void* d_out, int out_size, void* d_ws, size_t ws_size,
                              hipStream_t stream)
{
    const float* x_tx   = (const float*)d_in[0];
    const float* x_user = (const float*)d_in[1];
    const float* x_merch= (const float*)d_in[2];
    const float* Wp  = (const float*)d_in[3];
    const float* bp  = (const float*)d_in[4];
    const float* Wl  = (const float*)d_in[5];   // [2,4,256,256]
    const float* bl  = (const float*)d_in[6];   // [2,4,256]
    const float* Wr  = (const float*)d_in[7];   // [2,4,256,256]
    const float* Wc1 = (const float*)d_in[8];
    const float* bc1 = (const float*)d_in[9];
    const float* Wc2 = (const float*)d_in[10];
    const float* bc2 = (const float*)d_in[11];
    const float* Wv1 = (const float*)d_in[12];
    const float* bv1 = (const float*)d_in[13];
    const float* Wv2 = (const float*)d_in[14];
    const float* bv2 = (const float*)d_in[15];
    const int* e_ut_src = (const int*)d_in[16];
    const int* e_ut_dst = (const int*)d_in[17];
    const int* e_tm_src = (const int*)d_in[18];
    const int* e_tm_dst = (const int*)d_in[19];
    const int* e_mt_src = (const int*)d_in[20];
    const int* e_mt_dst = (const int*)d_in[21];
    const int* e_tu_src = (const int*)d_in[22];
    const int* e_tu_dst = (const int*)d_in[23];

    // Workspace layout (byte offsets, 256B aligned segments)
    char* w = (char*)d_ws;
    size_t o = 0;
    auto carve = [&](size_t bytes) { char* p = w + o; o += (bytes + 255) & ~(size_t)255; return p; };
    float*  acc_t  = (float*)carve((size_t)N_TX * HDIM * 4);
    float*  mean   = (float*)carve((size_t)N_TX * HDIM * 4);   // f32 atomic scratch; reused as head z
    float*  acc_u  = (float*)carve((size_t)N_USER * HDIM * 4);
    float*  acc_m  = (float*)carve((size_t)N_MERCH * HDIM * 4);
    float*  cnt    = (float*)carve((size_t)N_TX * 4);
    __bf16* xt_bf  = (__bf16*)carve((size_t)N_TX * HDIM * 2);
    __bf16* xu_bf  = (__bf16*)carve((size_t)N_USER * HDIM * 2);
    __bf16* xm_bf  = (__bf16*)carve((size_t)N_MERCH * HDIM * 2);
    __bf16* mbf_a  = (__bf16*)carve((size_t)N_TX * HDIM * 2);
    __bf16* mbf_b  = (__bf16*)carve((size_t)N_TX * HDIM * 2);
    (void)ws_size; (void)in_sizes; (void)n_in; (void)out_size;

    // Input projection + bf16 feature init
    proj_kernel<<<N_TX, 256, 0, stream>>>(x_tx, Wp, bp, xt_bf);
    f32_to_bf16_kernel<<<gblocks((long)N_USER * HDIM), 256, 0, stream>>>(x_user, xu_bf, (long)N_USER * HDIM);
    f32_to_bf16_kernel<<<gblocks((long)N_MERCH * HDIM), 256, 0, stream>>>(x_merch, xm_bf, (long)N_MERCH * HDIM);

    const size_t WSTEP = (size_t)HDIM * HDIM;
    for (int l = 0; l < 2; ++l) {
        const float* Wl_l = Wl + (size_t)l * 4 * WSTEP;
        const float* Wr_l = Wr + (size_t)l * 4 * WSTEP;
        const float* bl_l = bl + (size_t)l * 4 * HDIM;

        // Means into tx from both relations (both alive for the fused tx GEMMs)
        launch_segmean(stream, xu_bf, e_ut_src, e_ut_dst, N_TX, mean, cnt, mbf_a);   // user -> tx
        launch_segmean(stream, xm_bf, e_mt_src, e_mt_dst, N_TX, mean, cnt, mbf_b);   // merch -> tx

        // acc_t = (mbf_a@Wl0 + bl0 + xt@Wr0) + (mbf_b@Wl2 + bl2 + xt@Wr2)
        launch_gemm_fused_init(stream, mbf_a, Wl_l + 0 * WSTEP, xt_bf, Wr_l + 0 * WSTEP,
                               bl_l + 0 * HDIM, acc_t, N_TX, HDIM);
        launch_gemm_fused_acc(stream, mbf_b, Wl_l + 2 * WSTEP, xt_bf, Wr_l + 2 * WSTEP,
                              bl_l + 2 * HDIM, acc_t, N_TX, HDIM);

        // acc_m = mean(tx->merch)@Wl1 + bl1 + xm@Wr1
        launch_segmean(stream, xt_bf, e_tm_src, e_tm_dst, N_MERCH, mean, cnt, mbf_a);
        launch_gemm_fused_init(stream, mbf_a, Wl_l + 1 * WSTEP, xm_bf, Wr_l + 1 * WSTEP,
                               bl_l + 1 * HDIM, acc_m, N_MERCH, HDIM);

        // acc_u = mean(tx->user)@Wl3 + bl3 + xu@Wr3
        launch_segmean(stream, xt_bf, e_tu_src, e_tu_dst, N_USER, mean, cnt, mbf_a);
        launch_gemm_fused_init(stream, mbf_a, Wl_l + 3 * WSTEP, xu_bf, Wr_l + 3 * WSTEP,
                               bl_l + 3 * HDIM, acc_u, N_USER, HDIM);

        // HeteroConv mean + ReLU -> new bf16 features (all consumers above used old feats)
        relu_scale_kernel<<<gblocks((long)N_TX * HDIM), 256, 0, stream>>>(acc_t, xt_bf, 0.5f, (long)N_TX * HDIM);
        relu_scale_kernel<<<gblocks((long)N_MERCH * HDIM), 256, 0, stream>>>(acc_m, xm_bf, 1.0f, (long)N_MERCH * HDIM);
        relu_scale_kernel<<<gblocks((long)N_USER * HDIM), 256, 0, stream>>>(acc_u, xu_bf, 1.0f, (long)N_USER * HDIM);
    }

    // Heads (reuse `mean` f32 scratch as z: [N_TX,128])
    float* z = mean;
    float* out_f = (float*)d_out;
    launch_gemm_relu(stream, xt_bf, Wc1, bc1, z, N_TX, 128);
    head_out_kernel<<<(N_TX + 7) / 8, 256, 0, stream>>>(z, Wc2, bc2, out_f, N_TX);
    launch_gemm_relu(stream, xt_bf, Wv1, bv1, z, N_TX, 128);
    head_out_kernel<<<(N_TX + 7) / 8, 256, 0, stream>>>(z, Wv2, bv2, out_f + N_TX, N_TX);
}
